// TransformerXL_79087527788797
// MI455X (gfx1250) — matrix operand order repaired and checked by reading, not compiled, not run on hardware
//
#include <hip/hip_runtime.h>
#include <hip/hip_bf16.h>

// ---------------------------------------------------------------------------
// Transformer-XL forward (T=512, B=16, D=512, H=8, DH=64, DI=2048, L=4) for
// gfx1250 (MI455X).
//  - bf16 WMMA (f32 accumulate) for all GEMMs + attention AC / P*V
//  - async global->LDS (ASYNCcnt) double-buffered tile staging
//  - rel_shift in closed form: BD[i,j] = (q + r_r_bias) . rk[T + j - i]
//  - f32 softmax / layernorm
// ---------------------------------------------------------------------------

typedef __attribute__((ext_vector_type(16))) __bf16 v16bf;
typedef __attribute__((ext_vector_type(8)))  float  v8f;

#define WMMA_BF16(a, b, c) \
  __builtin_amdgcn_wmma_f32_16x16x32_bf16(false, (a), false, (b), (short)0, (c), false, false)

#define T_  512
#define B_  16
#define D_  512
#define H_  8
#define DH_ 64
#define DI_ 2048
#define L_  4

#define USE_ASYNC 1

#if __has_builtin(__builtin_amdgcn_s_wait_asynccnt)
#define WAIT_ASYNC(n) __builtin_amdgcn_s_wait_asynccnt(n)
#else
#define WAIT_ASYNC(n) asm volatile("s_wait_asynccnt %0" ::"i"(n) : "memory")
#endif

// Copy 16 contiguous bytes per lane from global to LDS via the async engine.
// VDST carries the per-lane LDS byte address, VADDR the 64-bit global address.
__device__ __forceinline__ void async_cp_b128(const float* gsrc, unsigned lds_byte) {
#if USE_ASYNC
  unsigned long long ga = (unsigned long long)(uintptr_t)gsrc;
  asm volatile("global_load_async_to_lds_b128 %0, %1, off"
               ::"v"(lds_byte), "v"(ga)
               : "memory");
#else
  // synchronous fallback: plain vector load + LDS store
  float4 v = *(const float4*)gsrc;
  *(float4*)((char*)nullptr + lds_byte) = v;  // unused when USE_ASYNC
#endif
}

// ---------------------------------------------------------------------------
// GEMM: C = act(A[MxK] * W[KxN] + bias). f32 in/out, bf16 WMMA core.
// Block = 256 thr (8 waves), tile 128(M) x 64(N), K-step 32, async double
// buffering on the K tiles. Requires M%128==0, N%64==0, K%32==0.
// FLAG: 0 = none, 1 = +bias, 2 = +bias then ReLU.
// ---------------------------------------------------------------------------
template <int FLAG>
__global__ __launch_bounds__(256)
void gemm_bf16_k(const float* __restrict__ A, const float* __restrict__ W,
                 const float* __restrict__ bias, float* __restrict__ C,
                 int N, int K) {
  constexpr int ASP = 36;  // f32 LDS row stride (144 B: 16B-aligned, bank-skewed)
  constexpr int WSP = 68;  // f32 LDS row stride (272 B)
  __shared__ float Asf[2][128 * ASP];
  __shared__ float Wsf[2][32 * WSP];

  const int tid  = threadIdx.x;
  const int lane = tid & 31;
  const int wv   = tid >> 5;
  const int hi   = lane >> 4;
  const int ln   = lane & 15;
  const int wm   = wv & 3;   // wave M block (32 rows)
  const int wn   = wv >> 2;  // wave N block (32 cols)
  const int bm   = blockIdx.y * 128;
  const int bn   = blockIdx.x * 64;
  const int nkt  = K >> 5;

  const unsigned ab0 = (unsigned)(uintptr_t)&Asf[0][0];
  const unsigned ab1 = (unsigned)(uintptr_t)&Asf[1][0];
  const unsigned wb0 = (unsigned)(uintptr_t)&Wsf[0][0];
  const unsigned wb1 = (unsigned)(uintptr_t)&Wsf[1][0];

  auto stage = [&](int buf, int kk) {
    unsigned ab = buf ? ab1 : ab0;
    unsigned wb = buf ? wb1 : wb0;
#pragma unroll
    for (int qq = 0; qq < 4; ++qq) {  // A tile: 128x32 f32 = 1024 16B chunks
      int ch = tid + 256 * qq;
      int r = ch >> 3, c = (ch & 7) << 2;
      async_cp_b128(A + (long)(bm + r) * K + kk + c,
                    ab + (unsigned)(r * ASP + c) * 4u);
    }
#pragma unroll
    for (int qq = 0; qq < 2; ++qq) {  // W tile: 32x64 f32 = 512 16B chunks
      int ch = tid + 256 * qq;
      int r = ch >> 4, c = (ch & 15) << 2;
      async_cp_b128(W + (long)(kk + r) * N + bn + c,
                    wb + (unsigned)(r * WSP + c) * 4u);
    }
  };

  v8f accs[2][2] = {{{}, {}}, {{}, {}}};

  stage(0, 0);
  for (int t = 0; t < nkt; ++t) {
    if (t + 1 < nkt) {
      stage((t + 1) & 1, (t + 1) * 32);
      WAIT_ASYNC(6);  // previous tile's 6 copies done; next tile in flight
    } else {
      WAIT_ASYNC(0);
    }
    __syncthreads();

    const float* As = Asf[t & 1];
    const float* Ws = Wsf[t & 1];
    const float* ap0 = As + (wm * 32 + ln) * ASP;
    const float* ap1 = ap0 + 16 * ASP;
    v16bf a0, a1, b0v, b1v;
#pragma unroll
    for (int i = 0; i < 8; ++i) {
      a0[i]     = (__bf16)ap0[hi * 16 + i];
      a0[i + 8] = (__bf16)ap0[hi * 16 + 8 + i];
      a1[i]     = (__bf16)ap1[hi * 16 + i];
      a1[i + 8] = (__bf16)ap1[hi * 16 + 8 + i];
    }
    const int c0 = wn * 32 + ln, c1 = c0 + 16;
#pragma unroll
    for (int i = 0; i < 16; ++i) {
      const float* wr = Ws + (hi * 16 + i) * WSP;
      b0v[i] = (__bf16)wr[c0];
      b1v[i] = (__bf16)wr[c1];
    }
    accs[0][0] = WMMA_BF16(a0, b0v, accs[0][0]);
    accs[0][1] = WMMA_BF16(a0, b1v, accs[0][1]);
    accs[1][0] = WMMA_BF16(a1, b0v, accs[1][0]);
    accs[1][1] = WMMA_BF16(a1, b1v, accs[1][1]);
    __syncthreads();
  }

#pragma unroll
  for (int ti = 0; ti < 2; ++ti)
#pragma unroll
    for (int tj = 0; tj < 2; ++tj) {
      int col = bn + wn * 32 + tj * 16 + ln;
      float bv = (FLAG != 0) ? bias[col] : 0.0f;
#pragma unroll
      for (int r = 0; r < 8; ++r) {
        int row = bm + wm * 32 + ti * 16 + hi * 8 + r;
        float v = accs[ti][tj][r] + bv;
        if (FLAG == 2) v = fmaxf(v, 0.0f);
        C[(long)row * N + col] = v;
      }
    }
}

// ---------------------------------------------------------------------------
// LayerNorm over D=512: y = (x [+res] - mean)*rsqrt(var+eps)*g + b.
// ---------------------------------------------------------------------------
__global__ __launch_bounds__(256)
void layernorm_k(const float* __restrict__ x, const float* __restrict__ res,
                 const float* __restrict__ g, const float* __restrict__ bta,
                 float* __restrict__ y, int transpose_in) {
  __shared__ float sS[8], sQ[8];
  const int row = blockIdx.x;
  const int tid = threadIdx.x;
  long base;
  if (transpose_in) {
    int t = row >> 4, b = row & 15;
    base = ((long)b * T_ + t) * D_;
  } else {
    base = (long)row * D_;
  }
  float v0 = x[base + tid];
  float v1 = x[base + tid + 256];
  if (res) {
    v0 += res[(long)row * D_ + tid];
    v1 += res[(long)row * D_ + tid + 256];
  }
  float s = v0 + v1;
  float q = v0 * v0 + v1 * v1;
#pragma unroll
  for (int o = 16; o > 0; o >>= 1) {
    s += __shfl_xor(s, o, 32);
    q += __shfl_xor(q, o, 32);
  }
  if ((tid & 31) == 0) { sS[tid >> 5] = s; sQ[tid >> 5] = q; }
  __syncthreads();
  float S = 0.0f, Q = 0.0f;
#pragma unroll
  for (int w = 0; w < 8; ++w) { S += sS[w]; Q += sQ[w]; }
  float mean = S * (1.0f / D_);
  float var  = Q * (1.0f / D_) - mean * mean;
  float rstd = rsqrtf(var + 1e-6f);
  y[(long)row * D_ + tid]       = (v0 - mean) * rstd * g[tid] + bta[tid];
  y[(long)row * D_ + tid + 256] = (v1 - mean) * rstd * g[tid + 256] + bta[tid + 256];
}

// ---------------------------------------------------------------------------
// Sinusoidal position embeddings.
// ---------------------------------------------------------------------------
__global__ __launch_bounds__(256)
void pos_emb_k(float* __restrict__ fw, float* __restrict__ bw) {
  const int j = blockIdx.x;   // 0..1023
  const int d = threadIdx.x;  // 0..255
  float invf = expf(-((2.0f * (float)d) / (float)D_) * logf(10000.0f));
  float pf = (float)(T_ - j);
  float pb = (float)(-T_ + j);
  float af = pf * invf, ab = pb * invf;
  fw[(long)j * D_ + d]       = sinf(af);
  fw[(long)j * D_ + 256 + d] = cosf(af);
  bw[(long)j * D_ + d]       = sinf(ab);
  bw[(long)j * D_ + 256 + d] = cosf(ab);
}

// ---------------------------------------------------------------------------
// Transpose-store internal (T*B, D) buffer to output (B, T, D) region.
// ---------------------------------------------------------------------------
__global__ __launch_bounds__(256)
void store_hid_k(const float* __restrict__ X, float* __restrict__ out) {
  const int r = blockIdx.x;  // t*B + b
  const int t = r >> 4, b = r & 15;
  long src = (long)r * D_;
  long dst = ((long)b * T_ + t) * D_;
  out[dst + threadIdx.x]       = X[src + threadIdx.x];
  out[dst + threadIdx.x + 256] = X[src + threadIdx.x + 256];
}

// ---------------------------------------------------------------------------
// Rel-attention for one (64-query tile, batch b, head n).
// Dynamic LDS ~188 KB (needs CDNA5's 320 KB WGP LDS).
// ---------------------------------------------------------------------------
__global__ __launch_bounds__(256)
void attn_k(const float* __restrict__ qkv, const float* __restrict__ rkfw,
            const float* __restrict__ rkbw, const float* __restrict__ rwbias,
            const float* __restrict__ rrbias, const unsigned char* __restrict__ seqmask,
            float* __restrict__ vec) {
  extern __shared__ char smem[];
  float*  qs   = (float*)smem;            // 64*64 f32   q + r_r_bias (for BD)
  __bf16* qb   = (__bf16*)(qs + 4096);    // 64*66 bf16  q + r_w_bias (for AC)
  float*  sc   = (float*)(qb + 4224);     // 64*516 f32  scores / probs
  float*  kf0  = sc + 64 * 516;           // 32*68 f32   K tile buf 0 (natural j,d)
  float*  kf1  = kf0 + 32 * 68;           // 32*68 f32   K tile buf 1
  float*  vf0  = kf1 + 32 * 68;           // 32*68 f32   V tile buf 0
  float*  vf1  = vf0 + 32 * 68;           // 32*68 f32   V tile buf 1
  float*  rwb  = vf1 + 32 * 68;           // 64
  float*  rrb  = rwb + 64;                // 64
  float*  linv = rrb + 64;                // 64

  const int tid  = threadIdx.x;
  const int lane = tid & 31;
  const int wv   = tid >> 5;
  const int hi   = lane >> 4;
  const int ln   = lane & 15;
  const int i0   = blockIdx.x * 64;
  const int b    = blockIdx.y;
  const int n    = blockIdx.z;
  const int tm   = wv & 3;   // 16-row tile of the 64 queries
  const int jh   = wv >> 2;  // 16-col half of a 32-key step

  // async stage of a 32x64 f32 K/V tile (natural [j][d] layout, stride 68)
  auto stage_kv = [&](float* dst, const float* gbase, int j0) {
    unsigned db = (unsigned)(uintptr_t)dst;
#pragma unroll
    for (int qq = 0; qq < 2; ++qq) {
      int ch = tid + 256 * qq;
      int jj = ch >> 4, c = (ch & 15) << 2;
      async_cp_b128(gbase + ((long)(j0 + jj) * B_ + b) * (3 * D_) + c,
                    db + (unsigned)(jj * 68 + c) * 4u);
    }
  };
  const float* kbase = qkv + D_ + n * DH_;      // K section, this head
  const float* vbase = qkv + 2 * D_ + n * DH_;  // V section, this head
  float* kf[2] = {kf0, kf1};
  float* vf[2] = {vf0, vf1};

  stage_kv(kf[0], kbase, 0);  // overlap first K tile with q staging

  if (tid < 64) {
    rwb[tid] = rwbias[n * DH_ + tid];
    rrb[tid] = rrbias[n * DH_ + tid];
  }
  __syncthreads();
  for (int e = tid; e < 64 * 64; e += 256) {
    int il = e >> 6, d = e & 63;
    float qv = qkv[((long)(i0 + il) * B_ + b) * (3 * D_) + n * DH_ + d];
    qs[il * 64 + d] = qv + rrb[d];            // pre-biased for BD
    qb[il * 66 + d] = (__bf16)(qv + rwb[d]);  // pre-biased for AC
  }
  __syncthreads();

  // ---- Phase 1: AC = (q+rwb) K^T, WMMA, async double-buffered K tiles ----
  for (int jt = 0; jt < 16; ++jt) {
    const int j0 = jt * 32;
    if (jt + 1 < 16) {
      stage_kv(kf[(jt + 1) & 1], kbase, j0 + 32);
      WAIT_ASYNC(2);
    } else {
      WAIT_ASYNC(0);
    }
    __syncthreads();
    const float* kt = kf[jt & 1];
    const __bf16* ap = &qb[(tm * 16 + ln) * 66];
    const float*  bp = &kt[(jh * 16 + ln) * 68];
    v8f acc = {};
#pragma unroll
    for (int k0 = 0; k0 < DH_; k0 += 32) {
      v16bf a, bb;
#pragma unroll
      for (int i = 0; i < 8; ++i) {
        a[i]     = ap[k0 + hi * 16 + i];
        a[i + 8] = ap[k0 + hi * 16 + 8 + i];
      }
#pragma unroll
      for (int i = 0; i < 16; ++i) bb[i] = (__bf16)bp[k0 + hi * 16 + i];
      acc = WMMA_BF16(a, bb, acc);
    }
#pragma unroll
    for (int r = 0; r < 8; ++r)
      sc[(tm * 16 + hi * 8 + r) * 516 + j0 + jh * 16 + ln] = acc[r];
    __syncthreads();
  }

  // ---- Phase 1b: += BD, closed-form rel_shift (Toeplitz band, VALU f32) ----
  const float* rk = (b < (B_ / 2) ? rkfw : rkbw) + n * DH_;
  for (int idx = tid; idx < 64 * T_; idx += 256) {
    int il = idx >> 9;
    int j  = idx & (T_ - 1);
    int u  = T_ + j - (i0 + il);  // in [1, 2T-1]
    const float4* rk4 = (const float4*)(rk + (long)u * (H_ * DH_));
    const float4* qr4 = (const float4*)&qs[il * 64];
    float s = 0.0f;
#pragma unroll
    for (int d4 = 0; d4 < 16; ++d4) {
      float4 qa = qr4[d4], rb = rk4[d4];
      s = fmaf(qa.x, rb.x, s);
      s = fmaf(qa.y, rb.y, s);
      s = fmaf(qa.z, rb.z, s);
      s = fmaf(qa.w, rb.w, s);
    }
    sc[il * 516 + j] += s;
  }

  stage_kv(vf[0], vbase, 0);  // overlap first V tile with softmax
  __syncthreads();

  // ---- Phase 2: masked, scaled softmax over keys (wave handles 8 rows) ----
  for (int rr = 0; rr < 8; ++rr) {
    int il = wv * 8 + rr;
    float* srow = &sc[il * 516];
    float m = -3.0e38f;
    for (int j = lane; j < T_; j += 32) {
      float s = srow[j] * 0.125f;                       // 1/sqrt(DH)
      if (seqmask[(long)b * T_ + j] == 0) s = -1.0e9f;  // key padding
      srow[j] = s;
      m = fmaxf(m, s);
    }
#pragma unroll
    for (int o = 16; o > 0; o >>= 1) m = fmaxf(m, __shfl_xor(m, o, 32));
    float l = 0.0f;
    for (int j = lane; j < T_; j += 32) {
      float p = __expf(srow[j] - m);
      srow[j] = p;
      l += p;
    }
#pragma unroll
    for (int o = 16; o > 0; o >>= 1) l += __shfl_xor(l, o, 32);
    if (lane == 0) linv[il] = 1.0f / l;
  }
  __syncthreads();

  // ---- Phase 3: out = P V, WMMA, async double-buffered V tiles ----
  const int dh = (wv >> 2) * 32;
  v8f o0 = {}, o1 = {};
  for (int jt = 0; jt < 16; ++jt) {
    const int j0 = jt * 32;
    if (jt + 1 < 16) {
      stage_kv(vf[(jt + 1) & 1], vbase, j0 + 32);
      WAIT_ASYNC(2);
    } else {
      WAIT_ASYNC(0);
    }
    __syncthreads();
    const float* vt = vf[jt & 1];
    const float* pr = &sc[(tm * 16 + ln) * 516 + j0];
    v16bf a;
#pragma unroll
    for (int i = 0; i < 8; ++i) {
      a[i]     = (__bf16)pr[hi * 16 + i];
      a[i + 8] = (__bf16)pr[hi * 16 + 8 + i];
    }
    v16bf b0v, b1v;
#pragma unroll
    for (int i = 0; i < 16; ++i) {
      const float* vr = vt + (hi * 16 + i) * 68;
      b0v[i] = (__bf16)vr[dh + ln];
      b1v[i] = (__bf16)vr[dh + 16 + ln];
    }
    o0 = WMMA_BF16(a, b0v, o0);
    o1 = WMMA_BF16(a, b1v, o1);
    __syncthreads();
  }
#pragma unroll
  for (int r = 0; r < 8; ++r) {
    int il = tm * 16 + hi * 8 + r;
    float sc1 = linv[il];
    long  grow = ((long)(i0 + il) * B_ + b) * D_ + n * DH_;
    vec[grow + dh + ln]      = o0[r] * sc1;
    vec[grow + dh + 16 + ln] = o1[r] * sc1;
  }
}

// ---------------------------------------------------------------------------
// Host driver
// ---------------------------------------------------------------------------
extern "C" void kernel_launch(void* const* d_in, const int* in_sizes, int n_in,
                              void* d_out, int out_size, void* d_ws, size_t ws_size,
                              hipStream_t stream) {
  (void)in_sizes; (void)n_in; (void)out_size; (void)ws_size;
  const float*         h        = (const float*)d_in[0];          // (B,T,D)
  const unsigned char* seq_mask = (const unsigned char*)d_in[1];  // (B,T) bool
  const float*         r_w_bias = (const float*)d_in[2];          // (H,DH)
  const float*         r_r_bias = (const float*)d_in[3];          // (H,DH)
  const float*         top_ln_g = (const float*)d_in[4];
  const float*         top_ln_b = (const float*)d_in[5];
  const float*         qkv_w    = (const float*)d_in[6];          // (L,D,3HD)
  const float*         r_net_w  = (const float*)d_in[7];          // (L,D,HD)
  const float*         o_w      = (const float*)d_in[8];          // (L,HD,D)
  const float*         attn_g   = (const float*)d_in[9];          // (L,D)
  const float*         attn_b   = (const float*)d_in[10];
  const float*         ff_w1    = (const float*)d_in[11];         // (L,D,DI)
  const float*         ff_b1    = (const float*)d_in[12];         // (L,DI)
  const float*         ff_w2    = (const float*)d_in[13];         // (L,DI,D)
  const float*         ff_b2    = (const float*)d_in[14];         // (L,D)
  const float*         ff_g     = (const float*)d_in[15];
  const float*         ff_b     = (const float*)d_in[16];
  float* out = (float*)d_out;
  float* ws  = (float*)d_ws;

  const long ROWS = (long)T_ * B_;  // 8192 internal rows
  const long NTOK = ROWS * D_;      // floats per (B,T,D) slab

  // workspace layout (floats); FFH aliases [QKV|VEC] (exactly 16.78M floats)
  float* X     = ws;
  float* Y     = X + NTOK;
  float* QKV   = Y + NTOK;
  float* VEC   = QKV + ROWS * 3 * D_;
  float* FFH   = QKV;  // alias
  float* POSFW = VEC + NTOK;
  float* POSBW = POSFW + (long)2 * T_ * D_;
  float* RKFW  = POSBW + (long)2 * T_ * D_;
  float* RKBW  = RKFW + (long)2 * T_ * D_;
  float* TMP   = RKBW + (long)2 * T_ * D_;

  const size_t ATTN_SMEM =
      4096 * 4 + 4224 * 2 + 64 * 516 * 4 + 4 * (32 * 68) * 4 + 3 * 64 * 4;  // 192512 B

  pos_emb_k<<<2 * T_, 256, 0, stream>>>(POSFW, POSBW);

  layernorm_k<<<ROWS, 256, 0, stream>>>(h, nullptr, top_ln_g, top_ln_b, X, 1);
  store_hid_k<<<ROWS, 256, 0, stream>>>(X, out + 1 * NTOK);  // hids[0]

  for (int l = 0; l < L_; ++l) {
    gemm_bf16_k<0><<<dim3(24, 64), 256, 0, stream>>>(
        X, qkv_w + (long)l * D_ * 3 * D_, nullptr, QKV, 3 * D_, D_);
    gemm_bf16_k<0><<<dim3(8, 8), 256, 0, stream>>>(
        POSFW, r_net_w + (long)l * D_ * D_, nullptr, RKFW, D_, D_);
    gemm_bf16_k<0><<<dim3(8, 8), 256, 0, stream>>>(
        POSBW, r_net_w + (long)l * D_ * D_, nullptr, RKBW, D_, D_);
    attn_k<<<dim3(T_ / 64, B_, H_), 256, ATTN_SMEM, stream>>>(
        QKV, RKFW, RKBW, r_w_bias, r_r_bias, seq_mask, VEC);
    gemm_bf16_k<0><<<dim3(8, 64), 256, 0, stream>>>(
        VEC, o_w + (long)l * D_ * D_, nullptr, TMP, D_, D_);
    layernorm_k<<<ROWS, 256, 0, stream>>>(TMP, X, attn_g + (long)l * D_,
                                          attn_b + (long)l * D_, Y, 0);
    gemm_bf16_k<2><<<dim3(32, 64), 256, 0, stream>>>(
        Y, ff_w1 + (long)l * D_ * DI_, ff_b1 + (long)l * DI_, FFH, DI_, D_);
    gemm_bf16_k<1><<<dim3(8, 64), 256, 0, stream>>>(
        FFH, ff_w2 + (long)l * DI_ * D_, ff_b2 + (long)l * D_, TMP, D_, DI_);
    layernorm_k<<<ROWS, 256, 0, stream>>>(TMP, Y, ff_g + (long)l * D_,
                                          ff_b + (long)l * D_, X, 0);
    store_hid_k<<<ROWS, 256, 0, stream>>>(X, out + (long)(2 + l) * NTOK);  // hids[l+1]
  }

  store_hid_k<<<ROWS, 256, 0, stream>>>(X, out);  // final out (== hids[L])
}
